// LSTM_8504035246509
// MI455X (gfx1250) — matrix operand order repaired and checked by reading, not compiled, not run on hardware
//
#include <hip/hip_runtime.h>
#include <hip/hip_bf16.h>

// ---- problem constants (B,S,I,H) = (128,512,512,1024) ----
#define B_ 128
#define S_ 512
#define I_ 512
#define H_ 1024
#define G_ 4096   // 4*H
#define K_ 1536   // I + H

// ---- GEMM tiling ----
#define BM 64     // batch rows per block (2 M-tiles over B=128)
#define BN 32     // permuted gate columns per block = 8 h-indices x 4 gates
#define KT 32     // WMMA K depth (bf16 16x16x32)
#define NC (K_ / KT)   // 48 K-chunks

typedef __attribute__((ext_vector_type(16))) __bf16 v16bf;
typedef __attribute__((ext_vector_type(8)))  float  v8f;
typedef __attribute__((ext_vector_type(4)))  float  f32x4;

__device__ __forceinline__ void async_stage_b128(const __bf16* src, unsigned ldsAddr) {
    unsigned long long ga = (unsigned long long)(size_t)src;
    asm volatile("global_load_async_to_lds_b128 %0, %1, off"
                 :: "v"(ldsAddr), "v"(ga) : "memory");
}
__device__ __forceinline__ void wait_async0() {
    asm volatile("s_wait_asynccnt 0x0" ::: "memory");
}
__device__ __forceinline__ void wait_ds0() {
    asm volatile("s_wait_dscnt 0x0" ::: "memory");
}

// ------------------------------------------------------------------
// One-time: WUt[n'][k] bf16, gate-interleaved columns: n' = h*4 + gate
// (gate 0..3 = f,i,g,o per reference slicing). k<I_ -> W, else U.
// ------------------------------------------------------------------
__global__ void cvt_weights(const float* __restrict__ W, const float* __restrict__ U,
                            __bf16* __restrict__ WUt) {
    int idx = blockIdx.x * blockDim.x + threadIdx.x;  // over G_*K_
    if (idx >= G_ * K_) return;
    int np = idx / K_;            // permuted column
    int k  = idx - np * K_;
    int hcol = np >> 2;           // h index
    int gate = np & 3;            // 0=f,1=i,2=g,3=o
    int n = gate * H_ + hcol;     // original gate column
    float v = (k < I_) ? W[(size_t)k * G_ + n] : U[(size_t)(k - I_) * G_ + n];
    WUt[idx] = (__bf16)v;
}

// biasP[n'] in the same permuted order
__global__ void permute_bias(const float* __restrict__ bias, float* __restrict__ biasP) {
    int np = blockIdx.x * blockDim.x + threadIdx.x;
    if (np >= G_) return;
    biasP[np] = bias[(np & 3) * H_ + (np >> 2)];
}

// Zero h (both buffers), c, and the output accumulator
__global__ void init_state(float* __restrict__ h0, float* __restrict__ h1,
                           float* __restrict__ c, float* __restrict__ out) {
    int idx = blockIdx.x * blockDim.x + threadIdx.x;
    if (idx < B_ * H_) { h0[idx] = 0.f; h1[idx] = 0.f; c[idx] = 0.f; out[idx] = 0.f; }
}

// ------------------------------------------------------------------
// Fused per-timestep kernel:
//   preacts(64x32 tile) = [x_t, h_in] @ WUt^T   (bf16 WMMA, fp32 acc)
//   then in-block LSTM cell update for its 64 (batch) x 8 (h) patch.
// Block: 128 threads = 4 waves. B tiles double-buffered in LDS via
// async-to-LDS so the fetch of chunk k+1 overlaps WMMA of chunk k.
// ------------------------------------------------------------------
__global__ void __launch_bounds__(128)
lstm_step_fused(const float* __restrict__ x, const float* __restrict__ hin,
                const __bf16* __restrict__ WUt, const float* __restrict__ biasP,
                float* __restrict__ c, float* __restrict__ hout,
                float* __restrict__ out, int t) {
    __shared__ __attribute__((aligned(32))) __bf16 Bsh[2][BN * KT];  // 2 x 2 KB
    __shared__ __attribute__((aligned(16))) float  Csh[BM * BN];     // 8 KB

    const int tid  = threadIdx.x;
    const int wave = tid >> 5;
    const int lane = tid & 31;
    const int half = lane >> 4;
    const int r    = lane & 15;

    const int nTiles = G_ / BN;                 // 128
    const int n0 = (blockIdx.x % nTiles) * BN;  // permuted-column base
    const int m0 = (blockIdx.x / nTiles) * BM;
    const int mw = m0 + wave * 16;
    const int row = mw + r;

    // cooperative B staging: 32 rows(n') x 32(k) bf16; each thread moves 16B
    const int sn = tid >> 2;      // 0..31
    const int sq = tid & 3;       // 16B quarter of a 64B row
    const unsigned ldsB0 = (unsigned)(size_t)&Bsh[0][sn * KT + sq * 8];
    const unsigned ldsB1 = (unsigned)(size_t)&Bsh[1][sn * KT + sq * 8];
    const __bf16* wrow = WUt + (size_t)(n0 + sn) * K_ + sq * 8;  // + kk per chunk

    const float* xrow = x + (size_t)row * (S_ * I_) + (size_t)t * I_;
    const float* hrow = hin + (size_t)row * H_;

    v8f acc0 = {};  // cols n0 .. n0+15
    v8f acc1 = {};  // cols n0+16 .. n0+31

    // prologue: stage chunk 0 into buffer 0
    async_stage_b128(wrow, ldsB0);
    wait_async0();
    __syncthreads();

    int buf = 0;
    for (int cidx = 0; cidx < NC; ++cidx) {
        const int kk = cidx * KT;
        // issue next chunk into the other buffer (overlaps with compute below)
        if (cidx + 1 < NC)
            async_stage_b128(wrow + kk + KT, buf ? ldsB0 : ldsB1);

        // A fragment: 16x32 bf16 from fp32 [x_t | h]; cvts co-execute with WMMA
        const float* arow = (kk < I_) ? (xrow + kk) : (hrow + (kk - I_));
        f32x4 p0 = *(const f32x4*)(arow + half * 16);
        f32x4 p1 = *(const f32x4*)(arow + half * 16 + 4);
        f32x4 p2 = *(const f32x4*)(arow + half * 16 + 8);
        f32x4 p3 = *(const f32x4*)(arow + half * 16 + 12);
        v16bf a;
#pragma unroll
        for (int e = 0; e < 4; ++e) {
            a[e]      = (__bf16)p0[e];
            a[4 + e]  = (__bf16)p1[e];
            a[8 + e]  = (__bf16)p2[e];
            a[12 + e] = (__bf16)p3[e];
        }

        const __bf16* bbase = &Bsh[buf][0];
        v16bf b0 = *(const v16bf*)&bbase[(0 * 16 + r) * KT + half * 16];
        v16bf b1 = *(const v16bf*)&bbase[(1 * 16 + r) * KT + half * 16];

        acc0 = __builtin_amdgcn_wmma_f32_16x16x32_bf16(false, a, false, b0,
                                                       (short)0, acc0, false, false);
        acc1 = __builtin_amdgcn_wmma_f32_16x16x32_bf16(false, a, false, b1,
                                                       (short)0, acc1, false, false);

        // retire this wave's LDS reads, then wait for next chunk's async write,
        // then block-wide barrier before switching buffers
        wait_ds0();
        wait_async0();
        __syncthreads();
        buf ^= 1;
    }

    // ---- exchange accumulators through LDS so one thread owns all 4 gates ----
    // D layout: VGPR v, lanes 0-15 -> M=v, lanes 16-31 -> M=8+v; N = lane%16
#pragma unroll
    for (int v = 0; v < 8; ++v) {
        int rl = (wave * 16) + v + 8 * half;          // local row 0..63
        Csh[rl * BN + r]      = acc0[v];
        Csh[rl * BN + 16 + r] = acc1[v];
    }
    __syncthreads();

    // ---- fused LSTM cell update: 64 rows x 8 h = 512 pairs, 4 per thread ----
#pragma unroll
    for (int i = 0; i < 4; ++i) {
        int p  = tid + 128 * i;
        int rl = p >> 3;                 // local row 0..63
        int hl = p & 7;                  // local h 0..7
        f32x4 q  = *(const f32x4*)&Csh[rl * BN + hl * 4];     // f,i,g,o preacts
        f32x4 bq = *(const f32x4*)&biasP[n0 + hl * 4];

        float fg = 1.f / (1.f + __expf(-(q[0] + bq[0])));
        float ig = 1.f / (1.f + __expf(-(q[1] + bq[1])));
        float gg = tanhf(q[2] + bq[2]);
        float og = 1.f / (1.f + __expf(-(q[3] + bq[3])));

        int hg  = (n0 >> 2) + hl;                       // global h index
        size_t idx = (size_t)(m0 + rl) * H_ + hg;
        float cn = fg * c[idx] + ig * gg;
        c[idx] = cn;
        float hn = og * tanhf(cn);
        hout[idx] = hn;
        out[idx] += hn;
    }
}

__global__ void finalize(float* __restrict__ out) {
    int idx = blockIdx.x * blockDim.x + threadIdx.x;
    if (idx < B_ * H_) out[idx] *= (1.0f / (float)S_);
}

// ------------------------------------------------------------------
extern "C" void kernel_launch(void* const* d_in, const int* in_sizes, int n_in,
                              void* d_out, int out_size, void* d_ws, size_t ws_size,
                              hipStream_t stream) {
    const float* x    = (const float*)d_in[0];   // (B,S,I)
    const float* W    = (const float*)d_in[1];   // (I,4H)
    const float* U    = (const float*)d_in[2];   // (H,4H)
    const float* bias = (const float*)d_in[3];   // (4H,)
    float* out = (float*)d_out;                  // (B,H)

    // workspace layout (~14.2 MB total)
    char* ws = (char*)d_ws;
    size_t off = 0;
    __bf16* WUt = (__bf16*)(ws + off);
    off += (size_t)G_ * K_ * sizeof(__bf16);            // 12.58 MB
    off = (off + 255) & ~(size_t)255;
    float* biasP = (float*)(ws + off); off += (size_t)G_ * sizeof(float);
    off = (off + 255) & ~(size_t)255;
    float* h0 = (float*)(ws + off); off += (size_t)B_ * H_ * sizeof(float);
    float* h1 = (float*)(ws + off); off += (size_t)B_ * H_ * sizeof(float);
    float* c  = (float*)(ws + off); off += (size_t)B_ * H_ * sizeof(float);

    {
        int total = G_ * K_;
        cvt_weights<<<(total + 255) / 256, 256, 0, stream>>>(W, U, WUt);
    }
    permute_bias<<<(G_ + 255) / 256, 256, 0, stream>>>(bias, biasP);
    init_state<<<(B_ * H_ + 255) / 256, 256, 0, stream>>>(h0, h1, c, out);

    dim3 grid((G_ / BN) * (B_ / BM));   // 128 * 2 = 256 blocks
    float* hA = h0;
    float* hB = h1;
    for (int t = 0; t < S_; ++t) {
        lstm_step_fused<<<grid, 128, 0, stream>>>(x, hA, WUt, biasP, c, hB, out, t);
        float* tmp = hA; hA = hB; hB = tmp;
    }
    finalize<<<(B_ * H_ + 255) / 256, 256, 0, stream>>>(out);
}
